// ContextAwareModel_12086037971477
// MI455X (gfx1250) — matrix operand order repaired and checked by reading, not compiled, not run on hardware
//
#include <hip/hip_runtime.h>
#include <hip/hip_bf16.h>

// ---------------------------------------------------------------------------
// Sizes (match reference)
#define TT   256      // seq len
#define BB   64       // batch
#define HH   256      // hidden
#define GG   1024     // 4*H gates
#define DD   300      // embedding dim
#define D_PAD 320     // D padded to multiple of 32 for WMMA K
#define K1   512      // layer-1 input dim (2H)
#define MM   (TT*BB)  // 16384 GEMM rows

typedef __attribute__((ext_vector_type(16))) _Float16 v16h;
typedef __attribute__((ext_vector_type(8)))  float    v8f;

// ---------------------------------------------------------------------------
// WMMA fragment loaders (CDNA5 ISA 7.12.2 layouts, wave32)
// A fragment: 16x32 (MxK), source row-major [row, ld] f16.
//   lane<16 : row = m0+lane,  K = {k0..k0+7} U {k0+16..k0+23}
//   lane>=16: row = m0+lane-16, K = {k0+8..k0+15} U {k0+24..k0+31}
__device__ __forceinline__ v16h load_frag_a(const _Float16* __restrict__ p, int ld, int lane) {
    const _Float16* q = p + (size_t)(lane & 15) * ld + ((lane >> 4) << 3);
    union { v16h v; float4 f[2]; } u;
    u.f[0] = *(const float4*)(q);
    u.f[1] = *(const float4*)(q + 16);
    return u.v;
}
// B fragment: 32x16 (KxN), sourced from W stored as NxK row-major (i.e. B = W^T).
//   lane<16 : col = n0+lane,  K = {k0..k0+7} U {k0+16..k0+23}
//   lane>=16: col = n0+lane-16, K = {k0+8..k0+15} U {k0+24..k0+31}
__device__ __forceinline__ v16h load_frag_b(const _Float16* __restrict__ p, int ld, int lane) {
    const _Float16* q = p + (size_t)(lane & 15) * ld + ((lane >> 4) << 3);
    union { v16h v; float4 f[2]; } u;
    u.f[0] = *(const float4*)(q);
    u.f[1] = *(const float4*)(q + 16);
    return u.v;
}

__device__ __forceinline__ float sigmoidf_(float x) { return 1.0f / (1.0f + __expf(-x)); }

// ---------------------------------------------------------------------------
// Weight convert / pad : dst[r][c] = (c<kin) ? f16(src[r][c]) : 0  (dst is rows x kout)
__global__ void cvt_pad_f16(const float* __restrict__ src, _Float16* __restrict__ dst,
                            long rows, int kin, int kout) {
    long idx = (long)blockIdx.x * blockDim.x + threadIdx.x;
    long total = rows * (long)kout;
    if (idx >= total) return;
    int  c = (int)(idx % kout);
    long r = idx / kout;
    dst[idx] = (c < kin) ? (_Float16)src[r * (long)kin + c] : (_Float16)0.0f;
}

__global__ void bias_sum(const float* __restrict__ a, const float* __restrict__ b,
                         float* __restrict__ d, int n) {
    int i = blockIdx.x * blockDim.x + threadIdx.x;
    if (i < n) d[i] = a[i] + b[i];
}

// ---------------------------------------------------------------------------
// Embedding gather -> x0[t*B+b][0..319] f16 (zero-padded cols 300..319)
__global__ void embed_pad(const int* __restrict__ tok, const float* __restrict__ emb,
                          _Float16* __restrict__ x0) {
    long idx = (long)blockIdx.x * blockDim.x + threadIdx.x;
    long total = (long)MM * D_PAD;
    if (idx >= total) return;
    int  c   = (int)(idx % D_PAD);
    long row = idx / D_PAD;                  // row = t*B + b
    int  t = (int)(row / BB), b = (int)(row % BB);
    if (c < DD) {
        int token = tok[b * TT + t];         // input_tensor is [B,T]
        x0[idx] = (_Float16)emb[(size_t)token * DD + c];
    } else {
        x0[idx] = (_Float16)0.0f;
    }
}

// ---------------------------------------------------------------------------
// Input projection GEMM: gates = A @ W^T + bias, stored FRAGMENT-MAJOR:
//   frag = (row_tile*4 + gate)*16 + colgrp, addr = frag*256 + lane*8 + r (f16),
//   matching the WMMA C/D register layout, so both the GEMM store and the
//   scan's accumulator load are one contiguous b128 per lane per tile.
// A: [MM,K] f16 row-major.  W: [2,GG,K] f16.  Xp: [2][MM*GG] f16 frag-major.
// block = 256 threads (8 waves), wave computes 16(M) x 64(N);
// grid = (MM/16, GG/512, 2).  K is compile-time -> fully unrolled k-loop.
template <int K>
__global__ __launch_bounds__(256) void xproj_gemm(
        const _Float16* __restrict__ A, const _Float16* __restrict__ W,
        const float* __restrict__ bias, _Float16* __restrict__ Xp) {
    const int dir  = blockIdx.z;
    const _Float16* Wd = W + (size_t)dir * GG * K;
    const float*    bd = bias + (size_t)dir * GG;
    _Float16*       Xd = Xp + (size_t)dir * MM * GG;

    const int wave = threadIdx.x >> 5;
    const int lane = threadIdx.x & 31;
    const int m0   = blockIdx.x * 16;
    const int n0   = blockIdx.y * 512 + wave * 64;
    const int nlo  = lane & 15;

    v8f acc[4];
#pragma unroll
    for (int j = 0; j < 4; ++j) {
        float bv = bd[n0 + j * 16 + nlo];
#pragma unroll
        for (int r = 0; r < 8; ++r) acc[j][r] = bv;
    }

#pragma unroll
    for (int kc = 0; kc < K; kc += 32) {
        v16h a = load_frag_a(A + (size_t)m0 * K + kc, K, lane);
#pragma unroll
        for (int j = 0; j < 4; ++j) {
            v16h b = load_frag_b(Wd + (size_t)(n0 + j * 16) * K + kc, K, lane);
            acc[j] = __builtin_amdgcn_wmma_f32_16x16x32_f16(
                         false, a, false, b, (short)0, acc[j], false, false);
        }
    }

    // fragment-major store: one contiguous 16B store per tile per lane
#pragma unroll
    for (int j = 0; j < 4; ++j) {
        const int n_tile = (n0 >> 4) + j;                       // 0..63
        const size_t frag = ((size_t)blockIdx.x * 4 + (n_tile >> 4)) * 16 + (n_tile & 15);
        union { float4 f4; _Float16 h[8]; } st;
#pragma unroll
        for (int r = 0; r < 8; ++r) st.h[r] = (_Float16)acc[j][r];
        *(float4*)(Xd + frag * 256 + (size_t)lane * 8) = st.f4;
    }
}

// ---------------------------------------------------------------------------
// Recurrent scan. Batch rows are independent in the recurrence, so:
// grid = 8 workgroups: blockIdx.x = dir*4 + batch-quad (16 rows each).
// block = 512 threads = 16 waves; wave w = hidden column-group w (16 cols).
// Per wave: 1 M-tile x 4 gate tiles; cell state stays in VGPRs across all T
// steps; h (16x256 f16) lives in LDS.  An opaque integer element-offset (empty
// asm, "+s") is added to the weight address each iteration so LICM cannot
// hoist the 32 B-fragments out of the t-loop (which would spill 256 VGPRs to
// scratch), while keeping pointer provenance intact so the loads stay
// global_load_b128 (not flat): weights stream from L2 every step, overlapped
// with the WMMAs.
__global__ __launch_bounds__(512) void lstm_scan(
        const _Float16* __restrict__ Xp,   // [2][T*B*G] f16, fragment-major
        const _Float16* __restrict__ whh,  // [2][G][H]  f16
        _Float16* __restrict__ out) {      // [T][B][2H] f16 (dir col offset)
    const int dir = blockIdx.x >> 2;
    const int mq  = blockIdx.x & 3;                  // batch quad: rows mq*16..+15
    const _Float16* Xd = Xp  + (size_t)dir * MM * GG;
    const _Float16* Wd = whh + (size_t)dir * GG * HH;

    __shared__ _Float16 hbuf[16][HH + 8];            // 16 x 264 f16 = 8.4 KB

    const int tid  = threadIdx.x;
    const int colg = tid >> 5;                       // wave id = column group 0..15
    const int lane = tid & 31;
    const int col0 = colg * 16;
    const int nlo  = lane & 15;
    const int rhi  = (lane >> 4) << 3;

    for (int i = tid; i < 16 * (HH + 8); i += blockDim.x)
        ((_Float16*)hbuf)[i] = (_Float16)0.0f;

    v8f cst;
#pragma unroll
    for (int r = 0; r < 8; ++r) cst[r] = 0.0f;
    __syncthreads();

    for (int t = 0; t < TT; ++t) {
        // opaque zero offset: defeats LICM of the weight loads while keeping
        // the pointer's global-address provenance (global_load, not flat)
        size_t wofs = 0;
        asm volatile("" : "+s"(wofs));
        const _Float16* Wt = Wd + wofs;

        const _Float16* Xt = Xd + (size_t)t * BB * GG;   // 256 frags * 256 halfs
        v8f acc[4];
        // accumulator init from fragment-major Xp: one b128 per tile per lane
#pragma unroll
        for (int g = 0; g < 4; ++g) {
            const size_t frag = ((size_t)mq * 4 + g) * 16 + colg;
            union { float4 f4; _Float16 h[8]; } ld;
            ld.f4 = *(const float4*)(Xt + frag * 256 + (size_t)lane * 8);
#pragma unroll
            for (int r = 0; r < 8; ++r) acc[g][r] = (float)ld.h[r];
        }
        // prefetch next timestep's fragments into cache while WMMAs run
        if (t + 1 < TT) {
#pragma unroll
            for (int g = 0; g < 4; ++g) {
                const size_t frag = ((size_t)mq * 4 + g) * 16 + colg;
                __builtin_prefetch(
                    (const void*)(Xt + (size_t)BB * GG + frag * 256 + (size_t)lane * 8),
                    0, 0);
            }
        }
        // recurrent GEMM: K = 256 -> 8 chunks of 32, A tile shared by 4 gates
#pragma unroll
        for (int kc = 0; kc < HH; kc += 32) {
            v16h a = load_frag_a(&hbuf[0][kc], HH + 8, lane);
#pragma unroll
            for (int g = 0; g < 4; ++g) {
                v16h b = load_frag_b(Wt + (size_t)(g * HH + col0) * HH + kc, HH, lane);
                acc[g] = __builtin_amdgcn_wmma_f32_16x16x32_f16(
                             false, a, false, b, (short)0, acc[g], false, false);
            }
        }
        __syncthreads();   // all waves done reading hbuf for step t
        // gate math (i,f,g,o) + write new h
#pragma unroll
        for (int r = 0; r < 8; ++r) {
            float ig = sigmoidf_(acc[0][r]);
            float fg = sigmoidf_(acc[1][r]);
            float gg = tanhf(acc[2][r]);
            float og = sigmoidf_(acc[3][r]);
            float cn = fg * cst[r] + ig * gg;
            cst[r] = cn;
            float hv = og * tanhf(cn);
            const int ml = r + rhi;                  // local row 0..15
            const int m  = mq * 16 + ml;             // batch row
            _Float16 h16 = (_Float16)hv;
            hbuf[ml][col0 + nlo] = h16;
            out[((size_t)t * BB + m) * (2 * HH) + dir * HH + col0 + nlo] = h16;
        }
        __syncthreads();   // new h visible to all waves before step t+1
    }
}

// ---------------------------------------------------------------------------
// Classifier: out[b] = sigmoid( out1[tgt[b], b, :] . clf_w + clf_b )
__global__ void classifier(const _Float16* __restrict__ out1, const int* __restrict__ tgt,
                           const float* __restrict__ w, const float* __restrict__ bcl,
                           float* __restrict__ o) {
    int b = threadIdx.x;
    if (b >= BB) return;
    int t = tgt[b];
    float acc = bcl[0];
    const _Float16* row = out1 + ((size_t)t * BB + b) * (2 * HH);
    for (int j = 0; j < 2 * HH; ++j) acc += (float)row[j] * w[j];
    o[b] = sigmoidf_(acc);
}

// ---------------------------------------------------------------------------
extern "C" void kernel_launch(void* const* d_in, const int* in_sizes, int n_in,
                              void* d_out, int out_size, void* d_ws, size_t ws_size,
                              hipStream_t stream) {
    const int*   tok   = (const int*)  d_in[0];   // [B,T]
    const int*   tgt   = (const int*)  d_in[1];   // [B]
    const float* emb   = (const float*)d_in[2];   // [V,D]
    const float* wih0  = (const float*)d_in[3];   // [2,G,D]
    const float* whh0  = (const float*)d_in[4];   // [2,G,H]
    const float* bih0  = (const float*)d_in[5];
    const float* bhh0  = (const float*)d_in[6];
    const float* wih1  = (const float*)d_in[7];   // [2,G,2H]
    const float* whh1  = (const float*)d_in[8];
    const float* bih1  = (const float*)d_in[9];
    const float* bhh1  = (const float*)d_in[10];
    const float* clf_w = (const float*)d_in[11];  // [1,2H]
    const float* clf_b = (const float*)d_in[12];  // [1]
    float* out = (float*)d_out;

    // workspace carve-out (256B aligned)
    char* ws = (char*)d_ws;
    size_t off = 0;
    auto carve = [&](size_t bytes) -> void* {
        off = (off + 255) & ~(size_t)255;
        void* p = ws + off;
        off += bytes;
        return p;
    };
    _Float16* x0     = (_Float16*)carve((size_t)MM * D_PAD * 2);        // 10.5 MB
    _Float16* wih0h  = (_Float16*)carve((size_t)2 * GG * D_PAD * 2);    // 1.3 MB
    _Float16* whh0h  = (_Float16*)carve((size_t)2 * GG * HH * 2);       // 1.0 MB
    _Float16* wih1h  = (_Float16*)carve((size_t)2 * GG * K1 * 2);       // 2.1 MB
    _Float16* whh1h  = (_Float16*)carve((size_t)2 * GG * HH * 2);       // 1.0 MB
    float*    bias0  = (float*)   carve((size_t)2 * GG * 4);
    float*    bias1  = (float*)   carve((size_t)2 * GG * 4);
    _Float16* Xp     = (_Float16*)carve((size_t)2 * MM * GG * 2);       // 64 MB (reused)
    _Float16* out0   = (_Float16*)carve((size_t)MM * 2 * HH * 2);       // 16.8 MB
    _Float16* out1   = (_Float16*)carve((size_t)MM * 2 * HH * 2);       // 16.8 MB
    (void)ws_size; (void)in_sizes; (void)n_in; (void)out_size;

    auto blocks = [](long total, int bs) { return (unsigned)((total + bs - 1) / bs); };

    // 1) weight conversion / padding + bias fusion
    cvt_pad_f16<<<blocks((long)2 * GG * D_PAD, 256), 256, 0, stream>>>(wih0, wih0h, 2L * GG, DD, D_PAD);
    cvt_pad_f16<<<blocks((long)2 * GG * HH, 256),   256, 0, stream>>>(whh0, whh0h, 2L * GG, HH, HH);
    cvt_pad_f16<<<blocks((long)2 * GG * K1, 256),   256, 0, stream>>>(wih1, wih1h, 2L * GG, K1, K1);
    cvt_pad_f16<<<blocks((long)2 * GG * HH, 256),   256, 0, stream>>>(whh1, whh1h, 2L * GG, HH, HH);
    bias_sum<<<blocks(2 * GG, 256), 256, 0, stream>>>(bih0, bhh0, bias0, 2 * GG);
    bias_sum<<<blocks(2 * GG, 256), 256, 0, stream>>>(bih1, bhh1, bias1, 2 * GG);

    // 2) embedding gather (f32 -> padded f16 [T*B, 320])
    embed_pad<<<blocks((long)MM * D_PAD, 256), 256, 0, stream>>>(tok, emb, x0);

    // 3) layer 0: time-parallel input projection (WMMA GEMM), then recurrent scan
    {
        dim3 g(MM / 16, GG / 512, 2);
        xproj_gemm<D_PAD><<<g, 256, 0, stream>>>(x0, wih0h, bias0, Xp);
    }
    lstm_scan<<<8, 512, 0, stream>>>(Xp, whh0h, out0);

    // 4) layer 1: same on the concatenated bidirectional output [T*B, 512]
    {
        dim3 g(MM / 16, GG / 512, 2);
        xproj_gemm<K1><<<g, 256, 0, stream>>>(out0, wih1h, bias1, Xp);
    }
    lstm_scan<<<8, 512, 0, stream>>>(Xp, whh1h, out1);

    // 5) gather at target timestep + logistic classifier
    classifier<<<1, 64, 0, stream>>>(out1, tgt, clf_w, clf_b, out);
}
